// HyenaOperator_27178553049160
// MI455X (gfx1250) — matrix operand order repaired and checked by reading, not compiled, not run on hardware
//
#include <hip/hip_runtime.h>
#include <math.h>

// ---------------------------------------------------------------------------
// Hyena forward for MI455X (gfx1250, wave32).
//   B=2, L=8192, D=768, F=768, ORDER=2, INNER=2304, n=2L=16384
// Pipeline:
//   k_cvt_f16      : u f32 -> f16 row-major
//   k_cvt_T        : weights f32 (K,N) -> f16 transposed (N,K)  [L2-resident]
//   k_filter       : pos-emb -> MLP -> decay-modulated h, stored (ORDER,F,2L)
//   k_gemm         : C = A(f16 MxK) * W^T(f16 NxK) + bias   (v_wmma 16x16x32)
//   k_shortconv_t  : depthwise conv(3) + transpose split -> vt/x1t/x2t (B,F,L)
//   k_fftconv  x2  : per-channel 16K FFT conv in LDS (real-packed v+ih trick)
//   k_cvt_T_vt     : vt (B,F,L) f32 -> vh (B*L,F) f16 for output GEMM
// ---------------------------------------------------------------------------

typedef __attribute__((ext_vector_type(16))) _Float16 v16h;
typedef __attribute__((ext_vector_type(8)))  _Float16 v8h;
typedef __attribute__((ext_vector_type(8)))  float    v8f;

#define NB   2
#define NL   8192
#define ND   768
#define NF   768
#define NINNER 2304
#define NFFT 16384

// ---------------- WMMA fragment loads (ISA 7.12.2, 16-bit, wave32) ----------
// A 16x32 fragment: lane m = lane&15, kh = lane>>4.
//   e=0..15 -> K = kh*16 + e (code's order, not ISA table; B same)       (32B run, two 16B loads)
// B 32x16 fragment from transposed WT (N,K): lane n, K = kh*16+e (32B run).
__device__ __forceinline__ v16h frag_2x8(const _Float16* p0, const _Float16* p1) {
  v8h lo = *(const v8h*)p0;
  v8h hi = *(const v8h*)p1;
  return __builtin_shufflevector(lo, hi, 0, 1, 2, 3, 4, 5, 6, 7,
                                         8, 9, 10, 11, 12, 13, 14, 15);
}

// C/D 16x16 f32: lane n = lane&15, M = (lane>>4)*8 + r
__device__ __forceinline__ void store_c(float* __restrict__ C, int ldc,
                                        int m0, int n0, const v8f& acc,
                                        const float* __restrict__ bias, int lane) {
  int nn = n0 + (lane & 15);
  int mh = lane >> 4;
  float badd = bias[nn];
#pragma unroll
  for (int r = 0; r < 8; ++r)
    C[(size_t)(m0 + mh * 8 + r) * ldc + nn] = acc[r] + badd;
}

// ---------------------------------------------------------------------------
__global__ __launch_bounds__(256) void k_cvt_f16(const float* __restrict__ src,
                                                 _Float16* __restrict__ dst, int n) {
  int i = blockIdx.x * blockDim.x + threadIdx.x;
  if (i < n) dst[i] = (_Float16)src[i];
}

// Transpose-convert: W (K x N) f32 -> WT (N x K) f16, tiled via LDS.
__global__ __launch_bounds__(256) void k_cvt_T(const float* __restrict__ src,
                                               _Float16* __restrict__ dst,
                                               int K, int N) {
  __shared__ float ts[32][33];
  int tx = threadIdx.x, ty = threadIdx.y;
  int k0 = blockIdx.x * 32, n0 = blockIdx.y * 32;
#pragma unroll
  for (int i = 0; i < 4; ++i)
    ts[ty + 8 * i][tx] = src[(size_t)(k0 + ty + 8 * i) * N + n0 + tx];
  __syncthreads();
#pragma unroll
  for (int i = 0; i < 4; ++i)
    dst[(size_t)(n0 + ty + 8 * i) * K + k0 + tx] = (_Float16)ts[tx][ty + 8 * i];
}

// Transpose-convert: vt (B, F, L) f32 -> vh (B*L, F) f16.
__global__ __launch_bounds__(256) void k_cvt_T_vt(const float* __restrict__ vt,
                                                  _Float16* __restrict__ vh) {
  __shared__ float ts[32][33];
  int tx = threadIdx.x, ty = threadIdx.y;
  int l0 = blockIdx.x * 32, f0 = blockIdx.y * 32, b = blockIdx.z;
#pragma unroll
  for (int i = 0; i < 4; ++i)  // read coalesced along L
    ts[ty + 8 * i][tx] = vt[((size_t)(b * NF + f0 + ty + 8 * i)) * NL + l0 + tx];
  __syncthreads();
#pragma unroll
  for (int i = 0; i < 4; ++i)  // write coalesced along F
    vh[((size_t)(b * NL + l0 + ty + 8 * i)) * NF + f0 + tx] =
        (_Float16)ts[tx][ty + 8 * i];
}

// ---------------------------------------------------------------------------
// Filter MLP + decay modulation. One block per time row r in [0, 2L).
// Output h_t layout: (ORDER, F, 2L) channel-major for contiguous FFT loads.
__global__ __launch_bounds__(256) void k_filter(
    const float* __restrict__ w1, const float* __restrict__ b1,
    const float* __restrict__ w2, const float* __restrict__ b2,
    const float* __restrict__ decay, float* __restrict__ h_t) {
  __shared__ float zsh[64];
  int r = blockIdx.x;
  int tid = threadIdx.x;
  float off = (r < NL) ? (float)r : (float)(r - NFFT);

  if (tid < 64) {
    float t[9];
    t[0] = off * (1.0f / 8192.0f);
    const float TWO_PI = 6.283185307179586f;
#pragma unroll
    for (int j = 0; j < 4; ++j) {
      float period = 4.0f + (float)j * ((8192.0f - 4.0f) / 3.0f);
      float ph = off * (TWO_PI / period);
      t[1 + j] = cosf(ph);          // |ph| up to ~1.3e4 -> precise libm
      t[5 + j] = sinf(ph);
    }
    float acc = b1[tid];
#pragma unroll
    for (int i = 0; i < 9; ++i) acc += t[i] * w1[i * 64 + tid];
    zsh[tid] = sinf(acc);
  }
  __syncthreads();

  float mod = (r < NL) ? (float)r * (1.0f / 8191.0f)
                       : (float)(NFFT - 1 - r) * (1.0f / 8191.0f);
#pragma unroll
  for (int q = 0; q < 6; ++q) {
    int c = tid + q * 256;                 // 0..1535
    float acc = b2[c];
    for (int j = 0; j < 64; ++j) acc += zsh[j] * w2[j * 1536 + c];
    int o = c / NF, f = c - o * NF;
    float d = fabsf(decay[o * NF + f]);
    h_t[((size_t)(o * NF + f) * NFFT) + r] = acc * expf(-mod * d);
  }
}

// ---------------------------------------------------------------------------
// Unified GEMM: C(M,N) = A(M,K f16 row-major) * WT(N,K f16)^T + bias.
// 256 threads = 8 waves in 2x4; wave tile 32x64 (2 A-frags x 4 B-frags,
// 8 v_wmma per K-step); block tile 64x256. All frag loads are b128.
// Six stream pointers advance by 64B/step -> no per-step base recompute.
__global__ __launch_bounds__(256) void k_gemm(const _Float16* __restrict__ A,
                                              const _Float16* __restrict__ WT,
                                              const float* __restrict__ bias,
                                              float* __restrict__ C,
                                              int M, int N, int K) {
  int tid = threadIdx.x, lane = tid & 31, wave = tid >> 5;
  int lm = lane & 15, kh = lane >> 4;
  int mw = blockIdx.x * 64 + (wave >> 2) * 32;
  int nw = blockIdx.y * 256 + (wave & 3) * 64;

  const _Float16* pa0 = A + (size_t)(mw + lm) * K + (kh << 4);
  const _Float16* pa1 = pa0 + (size_t)16 * K;
  const _Float16* pb0 = WT + (size_t)(nw + lm) * K + (kh << 4);
  const _Float16* pb1 = pb0 + (size_t)16 * K;
  const _Float16* pb2 = pb0 + (size_t)32 * K;
  const _Float16* pb3 = pb0 + (size_t)48 * K;

  v8f z = {0.f, 0.f, 0.f, 0.f, 0.f, 0.f, 0.f, 0.f};
  v8f acc[2][4] = {{z, z, z, z}, {z, z, z, z}};

  for (int k0 = 0; k0 < K; k0 += 32) {
    v16h a0 = frag_2x8(pa0, pa0 + 8);
    v16h a1 = frag_2x8(pa1, pa1 + 8);
    v16h b0 = frag_2x8(pb0, pb0 + 8);
    v16h b1 = frag_2x8(pb1, pb1 + 8);
    v16h b2 = frag_2x8(pb2, pb2 + 8);
    v16h b3 = frag_2x8(pb3, pb3 + 8);
    if (k0 + 64 < K) {   // prefetch HBM-streamed A two tiles ahead
      __builtin_prefetch(pa0 + 64, 0, 1);
      __builtin_prefetch(pa1 + 64, 0, 1);
    }
    acc[0][0] = __builtin_amdgcn_wmma_f32_16x16x32_f16(false, a0, false, b0, (short)0, acc[0][0], false, false);
    acc[0][1] = __builtin_amdgcn_wmma_f32_16x16x32_f16(false, a0, false, b1, (short)0, acc[0][1], false, false);
    acc[0][2] = __builtin_amdgcn_wmma_f32_16x16x32_f16(false, a0, false, b2, (short)0, acc[0][2], false, false);
    acc[0][3] = __builtin_amdgcn_wmma_f32_16x16x32_f16(false, a0, false, b3, (short)0, acc[0][3], false, false);
    acc[1][0] = __builtin_amdgcn_wmma_f32_16x16x32_f16(false, a1, false, b0, (short)0, acc[1][0], false, false);
    acc[1][1] = __builtin_amdgcn_wmma_f32_16x16x32_f16(false, a1, false, b1, (short)0, acc[1][1], false, false);
    acc[1][2] = __builtin_amdgcn_wmma_f32_16x16x32_f16(false, a1, false, b2, (short)0, acc[1][2], false, false);
    acc[1][3] = __builtin_amdgcn_wmma_f32_16x16x32_f16(false, a1, false, b3, (short)0, acc[1][3], false, false);
    pa0 += 32; pa1 += 32;
    pb0 += 32; pb1 += 32; pb2 += 32; pb3 += 32;
  }
#pragma unroll
  for (int i = 0; i < 2; ++i)
#pragma unroll
    for (int j = 0; j < 4; ++j)
      store_c(C, N, mw + 16 * i, nw + 16 * j, acc[i][j], bias, lane);
}

// ---------------------------------------------------------------------------
// Depthwise width-3 conv (SAME) fused with transpose-split into (B,F,L).
__global__ __launch_bounds__(256) void k_shortconv_t(
    const float* __restrict__ up, const float* __restrict__ sw,
    const float* __restrict__ sb, float* __restrict__ vt,
    float* __restrict__ x1t, float* __restrict__ x2t) {
  __shared__ float ts[32][33];
  int tx = threadIdx.x, ty = threadIdx.y;
  int lbase = blockIdx.x * 32;
  int cbase = blockIdx.y * 32;
  int b = blockIdx.z;
  int c = cbase + tx;
  float w0 = sw[c], w1 = sw[NINNER + c], w2 = sw[2 * NINNER + c];
  float bb = sb[c];
  const float* ub = up + (size_t)b * NL * NINNER;
#pragma unroll
  for (int i = 0; i < 4; ++i) {
    int ll = ty + i * 8;
    int l = lbase + ll;
    float um  = (l > 0)      ? ub[(size_t)(l - 1) * NINNER + c] : 0.0f;
    float u0  =                ub[(size_t)l       * NINNER + c];
    float upx = (l < NL - 1) ? ub[(size_t)(l + 1) * NINNER + c] : 0.0f;
    ts[ll][tx] = w0 * um + w1 * u0 + w2 * upx + bb;
  }
  __syncthreads();
  int sel = cbase / NF;                       // tile lies in exactly one split
  float* obuf = (sel == 0) ? vt : (sel == 1) ? x1t : x2t;
  int cw0 = cbase - sel * NF;
#pragma unroll
  for (int i = 0; i < 4; ++i) {
    int cl = ty + i * 8;
    obuf[((size_t)(b * NF + cw0 + cl)) * NL + lbase + tx] = ts[tx][cl];
  }
}

// ---------------------------------------------------------------------------
// In-LDS radix-2 DIF sweep (natural -> bit-reversed). sgn=-1 fwd, +1 inverse.
__device__ __forceinline__ void fft_dif(float2* S, int tid, float sgn) {
  for (int hb = 13; hb >= 0; --hb) {
    int half = 1 << hb;
    for (int g = tid; g < NFFT / 2; g += 256) {
      int t  = g & (half - 1);
      int i0 = ((g >> hb) << (hb + 1)) + t;
      int i1 = i0 + half;
      float ang = sgn * 3.14159265358979f * ((float)t / (float)half);
      float sn, cs;
      __sincosf(ang, &sn, &cs);
      float2 a = S[i0], b = S[i1];
      S[i0] = make_float2(a.x + b.x, a.y + b.y);
      float dx = a.x - b.x, dy = a.y - b.y;
      S[i1] = make_float2(dx * cs - dy * sn, dx * sn + dy * cs);
    }
    __syncthreads();
  }
}

// One block per (b,f) channel. LDS: 16384 complex = 128 KB (dynamic).
// Real-packing: S = v + i*h -> single forward FFT covers both spectra.
__global__ __launch_bounds__(256) void k_fftconv(float* __restrict__ vt,
                                                 const float* __restrict__ xt,
                                                 const float* __restrict__ hbase,
                                                 const float* __restrict__ bias_o) {
  extern __shared__ float2 S[];
  int ch = blockIdx.x;                 // b*F + f
  int f  = ch % NF;
  int tid = threadIdx.x;
  float* v = vt + (size_t)ch * NL;
  const float* x = xt + (size_t)ch * NL;
  const float* h = hbase + (size_t)f * NFFT;

  for (int i = tid; i < NFFT; i += 256) {
    float vr = (i < NL) ? v[i] : 0.0f;   // zero-pad v to 2L
    S[i] = make_float2(vr, h[i]);        // h spans full 2L (no pad)
  }
  __syncthreads();

  fft_dif(S, tid, -1.0f);                // forward, output bit-reversed

  for (int i = tid; i < NFFT; i += 256) { // bit-reversal permute -> natural
    int r = __brev((unsigned)i) >> 18;
    if (r > i) { float2 tmp = S[i]; S[i] = S[r]; S[r] = tmp; }
  }
  __syncthreads();

  const float inv_n = 1.0f / (float)NFFT;
  if (tid == 0) {                        // DC and Nyquist bins (real)
    float2 s0 = S[0];
    S[0]  = make_float2(s0.x * s0.y * inv_n, 0.0f);
    float2 sm = S[NL];
    S[NL] = make_float2(sm.x * sm.y * inv_n, 0.0f);
  }
  for (int k = 1 + tid; k < NL; k += 256) {
    int nk = NFFT - k;
    float2 sk = S[k], sc = S[nk];
    float Vr = 0.5f * (sk.x + sc.x);     // V = (S[k]+conj(S[n-k]))/2
    float Vi = 0.5f * (sk.y - sc.y);
    float Hr = 0.5f * (sk.y + sc.y);     // H = (S[k]-conj(S[n-k]))/(2i)
    float Hi = -0.5f * (sk.x - sc.x);
    float Pr = (Vr * Hr - Vi * Hi) * inv_n;
    float Pi = (Vr * Hi + Vi * Hr) * inv_n;
    S[k]  = make_float2(Pr,  Pi);
    S[nk] = make_float2(Pr, -Pi);        // Hermitian symmetry
  }
  __syncthreads();

  fft_dif(S, tid, +1.0f);                // unscaled inverse, bit-reversed out

  float bv = bias_o[f];
  for (int l = tid; l < NL; l += 256) {
    int rl = __brev((unsigned)l) >> 18;
    float y = S[rl].x;
    float res = (y + v[l] * bv) * x[l];
    v[l] = res;
  }
}

// ---------------------------------------------------------------------------
extern "C" void kernel_launch(void* const* d_in, const int* in_sizes, int n_in,
                              void* d_out, int out_size, void* d_ws, size_t ws_size,
                              hipStream_t stream) {
  const float* u     = (const float*)d_in[0];
  const float* fw1   = (const float*)d_in[1];
  const float* fb1   = (const float*)d_in[2];
  const float* fw2   = (const float*)d_in[3];
  const float* fb2   = (const float*)d_in[4];
  const float* decay = (const float*)d_in[5];
  const float* bias  = (const float*)d_in[6];
  const float* inw   = (const float*)d_in[7];
  const float* inb   = (const float*)d_in[8];
  const float* shw   = (const float*)d_in[9];
  const float* shb   = (const float*)d_in[10];
  const float* outw  = (const float*)d_in[11];
  const float* outb  = (const float*)d_in[12];
  float* out = (float*)d_out;

  // workspace carve-out (~458 MB total)
  char* p = (char*)d_ws;
  auto carve = [&](size_t bytes) {
    char* q = p;
    p += (bytes + 255) & ~(size_t)255;
    return q;
  };
  float*    uproj = (float*)   carve((size_t)NB * NL * NINNER * 4);  // 151 MB
  float*    h_t   = (float*)   carve((size_t)2 * NF * NFFT * 4);     // 101 MB
  float*    vt    = (float*)   carve((size_t)NB * NF * NL * 4);      //  50 MB
  float*    x1t   = (float*)   carve((size_t)NB * NF * NL * 4);
  float*    x2t   = (float*)   carve((size_t)NB * NF * NL * 4);
  _Float16* uh    = (_Float16*)carve((size_t)NB * NL * ND * 2);      //  25 MB
  _Float16* w1hT  = (_Float16*)carve((size_t)NINNER * ND * 2);       // (N,K)
  _Float16* w2hT  = (_Float16*)carve((size_t)ND * NF * 2);           // (N,K)
  _Float16* vh    = (_Float16*)carve((size_t)NB * NL * NF * 2);      //  25 MB

  const int n_u = NB * NL * ND;
  k_cvt_f16<<<(n_u + 255) / 256, 256, 0, stream>>>(u, uh, n_u);

  // weights: (K,N) f32 -> (N,K) f16 so B-fragments are contiguous b128 loads
  k_cvt_T<<<dim3(ND / 32, NINNER / 32), dim3(32, 8), 0, stream>>>(inw, w1hT, ND, NINNER);
  k_cvt_T<<<dim3(NF / 32, ND / 32), dim3(32, 8), 0, stream>>>(outw, w2hT, NF, ND);

  k_filter<<<NFFT, 256, 0, stream>>>(fw1, fb1, fw2, fb2, decay, h_t);

  // in_proj: (16384 x 768) @ (768 x 2304)
  k_gemm<<<dim3((NB * NL) / 64, NINNER / 256), 256, 0, stream>>>(
      uh, w1hT, inb, uproj, NB * NL, NINNER, ND);

  k_shortconv_t<<<dim3(NL / 32, NINNER / 32, NB), dim3(32, 8), 0, stream>>>(
      uproj, shw, shb, vt, x1t, x2t);

  // two Hyena orders; 128 KB dynamic LDS (fits 2 WGs per 320 KB WGP)
  k_fftconv<<<NB * NF, 256, 131072, stream>>>(vt, x1t, h_t, bias);
  k_fftconv<<<NB * NF, 256, 131072, stream>>>(vt, x2t, h_t + (size_t)NF * NFFT,
                                              bias + NF);

  // vt (B,F,L) f32 -> vh (B*L,F) f16 so output GEMM A-frags are b128 loads
  k_cvt_T_vt<<<dim3(NL / 32, NF / 32, NB), dim3(32, 8), 0, stream>>>(vt, vh);

  // out_proj: (16384 x 768) @ (768 x 768), rows match d_out (B,L,D) exactly
  k_gemm<<<dim3((NB * NL) / 64, ND / 256), 256, 0, stream>>>(
      vh, w2hT, outb, out, NB * NL, ND, NF);
}
